// MultiHeadAttention_79731772882970
// MI455X (gfx1250) — matrix operand order repaired and checked by reading, not compiled, not run on hardware
//
#include <hip/hip_runtime.h>
#include <hip/hip_bf16.h>

typedef _Float16 f16;
typedef __attribute__((ext_vector_type(16))) _Float16 v16h;
typedef __attribute__((ext_vector_type(8)))  _Float16 v8h;
typedef __attribute__((ext_vector_type(4)))  _Float16 v4h;
typedef __attribute__((ext_vector_type(8)))  float    v8f;
typedef __attribute__((ext_vector_type(4)))  float    v4f;

__device__ __forceinline__ v16h make_v16(v8h lo, v8h hi) {
  v16h r;
#pragma unroll
  for (int i = 0; i < 8; ++i) { r[i] = lo[i]; r[i + 8] = hi[i]; }
  return r;
}

__device__ __forceinline__ v4h cvt4(v4f f) {
  v4h h;
#pragma unroll
  for (int i = 0; i < 4; ++i) h[i] = (f16)f[i];
  return h;
}

// ---------------------------------------------------------------------------
// Kernel 1: QKV projection.  Y = x @ w_qkv^T + b_qkv, split into q/k/v f16
// buffers with [B,H,T,HD] layout.  M=16384, N=768, K=256.
// Block: 256 threads (8 waves), tile 128(M) x 64(N), K-step 32.
// ---------------------------------------------------------------------------
__global__ __launch_bounds__(256) void qkv_proj_kernel(
    const float* __restrict__ x, const float* __restrict__ w_qkv,
    const float* __restrict__ b_qkv,
    f16* __restrict__ q_ws, f16* __restrict__ k_ws, f16* __restrict__ v_ws) {
  __shared__ f16 As[128 * 40];
  __shared__ f16 Bs[64 * 40];

  const int tid = threadIdx.x;
  const int lane = tid & 31;
  const int wv = tid >> 5;     // wave 0..7
  const int wm = wv & 3;       // M sub-block (32 rows each)
  const int wn = wv >> 2;      // N sub-block (32 cols each)
  const int lo = lane & 15, hi = lane >> 4;
  const int m_tile = blockIdx.y * 128;
  const int n_tile = blockIdx.x * 64;

  v8f acc[2][2] = {};

  for (int k0 = 0; k0 < 256; k0 += 32) {
    // stage A tile 128x32 (f32 -> f16), float4 loads, b64 LDS stores
#pragma unroll
    for (int e = 0; e < 4; ++e) {
      int qidx = tid + 256 * e;              // 0..1023 quads
      int r = qidx >> 3, c4 = (qidx & 7) * 4;
      v4f f = *(const v4f*)&x[(size_t)(m_tile + r) * 256 + k0 + c4];
      *(v4h*)&As[r * 40 + c4] = cvt4(f);
    }
    // stage B tile 64x32: Bwmma[k][n] = w_qkv[n][k]
#pragma unroll
    for (int e = 0; e < 2; ++e) {
      int qidx = tid + 256 * e;              // 0..511 quads
      int r = qidx >> 3, c4 = (qidx & 7) * 4;
      v4f f = *(const v4f*)&w_qkv[(size_t)(n_tile + r) * 256 + k0 + c4];
      *(v4h*)&Bs[r * 40 + c4] = cvt4(f);
    }
    __syncthreads();

    v16h a[2], b[2];
#pragma unroll
    for (int i = 0; i < 2; ++i) {
      const f16* ap = &As[(wm * 32 + i * 16 + lo) * 40 + hi * 8];
      a[i] = make_v16(*(const v8h*)ap, *(const v8h*)(ap + 16));
    }
#pragma unroll
    for (int jn = 0; jn < 2; ++jn) {
      const f16* bp = &Bs[(wn * 32 + jn * 16 + lo) * 40 + hi * 8];
      b[jn] = make_v16(*(const v8h*)bp, *(const v8h*)(bp + 16));
    }
#pragma unroll
    for (int i = 0; i < 2; ++i)
#pragma unroll
      for (int jn = 0; jn < 2; ++jn)
        acc[i][jn] = __builtin_amdgcn_wmma_f32_16x16x32_f16(
            false, a[i], false, b[jn], (short)0, acc[i][jn], false, false);
    __syncthreads();
  }

  // bias + scatter into q/k/v f16 [B,H,T,HD]
#pragma unroll
  for (int jn = 0; jn < 2; ++jn) {
    int n = n_tile + wn * 32 + jn * 16 + lo;  // 0..767
    float bias = b_qkv[n];
    int sel = n >> 8;
    int rem = n & 255;
    int h = rem >> 6, d = rem & 63;
    f16* dst = (sel == 0) ? q_ws : (sel == 1) ? k_ws : v_ws;
#pragma unroll
    for (int i = 0; i < 2; ++i) {
      int mrow = m_tile + wm * 32 + i * 16;
#pragma unroll
      for (int r = 0; r < 8; ++r) {
        int m = mrow + r + hi * 8;
        int bb = m >> 11, t = m & 2047;
        dst[(size_t)(((bb * 4 + h) * 2048) + t) * 64 + d] =
            (f16)(acc[i][jn][r] + bias);
      }
    }
  }
}

// ---------------------------------------------------------------------------
// Kernel 2: flash attention per (b,h).  Block = 128 query rows; each wave
// owns 16 rows.  KV streamed through LDS in chunks of 64 keys.
// Output: attn f16 in [B,T,C] layout (ready to be GEMM A-matrix).
// ---------------------------------------------------------------------------
__global__ __launch_bounds__(256) void attn_kernel(
    const f16* __restrict__ q_ws, const f16* __restrict__ k_ws,
    const f16* __restrict__ v_ws, f16* __restrict__ attn_ws) {
  __shared__ f16 Ks[64 * 72];    // [key][d], padded
  __shared__ f16 Vt[64 * 72];    // [d][key], padded (transposed V)
  __shared__ f16 Ps[8 * 1024];   // per-wave P tile 16x64

  const int tid = threadIdx.x;
  const int lane = tid & 31;
  const int wv = tid >> 5;
  const int lo = lane & 15, hi = lane >> 4;
  const int bh = blockIdx.y;                 // 0..31  (b*4 + h)
  const int qrow0 = blockIdx.x * 128 + wv * 16;

  const f16* qb = q_ws + (size_t)bh * 2048 * 64;
  const f16* kb = k_ws + (size_t)bh * 2048 * 64;
  const f16* vb = v_ws + (size_t)bh * 2048 * 64;

  // Q A-fragments (d = 0..31, 32..63), resident for the whole kernel
  v16h aQ[2];
  {
    const f16* qp = qb + (size_t)(qrow0 + lo) * 64;
#pragma unroll
    for (int fb = 0; fb < 2; ++fb) {
      const f16* p = qp + fb * 32 + hi * 8;
      aQ[fb] = make_v16(*(const v8h*)p, *(const v8h*)(p + 16));
    }
  }

  v8f acc[4] = {};
  float m_r[8], l_r[8];
#pragma unroll
  for (int r = 0; r < 8; ++r) { m_r[r] = -1e30f; l_r[r] = 0.f; }
  const float scale = 0.125f;  // 1/sqrt(64)

  const int ldr = tid >> 2;          // 0..63 : kv row this thread stages
  const int ldc = (tid & 3) * 16;    // 0,16,32,48 : d-segment (16 halves)

  for (int kv0 = 0; kv0 < 2048; kv0 += 64) {
    // stage K (row-major) and V (transposed) chunk of 64 keys
    {
      const f16* ksrc = kb + (size_t)(kv0 + ldr) * 64 + ldc;
      *(v8h*)&Ks[ldr * 72 + ldc]     = *(const v8h*)ksrc;
      *(v8h*)&Ks[ldr * 72 + ldc + 8] = *(const v8h*)(ksrc + 8);
      const f16* vsrc = vb + (size_t)(kv0 + ldr) * 64 + ldc;
      v8h v0 = *(const v8h*)vsrc;
      v8h v1 = *(const v8h*)(vsrc + 8);
#pragma unroll
      for (int i = 0; i < 8; ++i) {
        Vt[(ldc + i) * 72 + ldr]     = v0[i];
        Vt[(ldc + 8 + i) * 72 + ldr] = v1[i];
      }
    }
    __syncthreads();

    // scores S[16 x 64] = Q(16x64) . K^T : four 16-wide key tiles
    v8f sc[4];
#pragma unroll
    for (int t = 0; t < 4; ++t) {
      v8f z = {};
#pragma unroll
      for (int s = 0; s < 2; ++s) {
        const f16* p = &Ks[(t * 16 + lo) * 72 + s * 32 + hi * 8];
        v16h bK = make_v16(*(const v8h*)p, *(const v8h*)(p + 16));
        z = __builtin_amdgcn_wmma_f32_16x16x32_f16(false, aQ[s], false, bK,
                                                   (short)0, z, false, false);
      }
      sc[t] = z;
    }

    // online softmax (rows live across the 16-lane half groups)
    f16* Pw = &Ps[wv * 1024];
#pragma unroll
    for (int r = 0; r < 8; ++r) {
      float sv[4];
#pragma unroll
      for (int t = 0; t < 4; ++t) sv[t] = sc[t][r] * scale;
      float mx = fmaxf(fmaxf(sv[0], sv[1]), fmaxf(sv[2], sv[3]));
#pragma unroll
      for (int off = 8; off >= 1; off >>= 1)
        mx = fmaxf(mx, __shfl_xor(mx, off, 32));
      float mnew = fmaxf(m_r[r], mx);
      float alpha = __expf(m_r[r] - mnew);
      float pv[4];
#pragma unroll
      for (int t = 0; t < 4; ++t) pv[t] = __expf(sv[t] - mnew);
      float rs = (pv[0] + pv[1]) + (pv[2] + pv[3]);
#pragma unroll
      for (int off = 8; off >= 1; off >>= 1)
        rs += __shfl_xor(rs, off, 32);
      l_r[r] = l_r[r] * alpha + rs;
      m_r[r] = mnew;
#pragma unroll
      for (int t = 0; t < 4; ++t) acc[t][r] *= alpha;
      int prow = r + hi * 8;
#pragma unroll
      for (int t = 0; t < 4; ++t)
        Pw[prow * 64 + t * 16 + lo] = (f16)pv[t];
    }
    // C-layout -> A-frag layout goes through LDS; order within our own wave
    asm volatile("s_wait_dscnt 0" ::: "memory");

    v16h aP[2];
#pragma unroll
    for (int fb = 0; fb < 2; ++fb) {
      const f16* pp = &Pw[lo * 64 + fb * 32 + hi * 8];
      aP[fb] = make_v16(*(const v8h*)pp, *(const v8h*)(pp + 16));
    }

    // O += P(16x64) @ V(64x64) : four 16-wide d tiles, two kk steps each
#pragma unroll
    for (int t = 0; t < 4; ++t) {
      v8f z = acc[t];
#pragma unroll
      for (int fb = 0; fb < 2; ++fb) {
        const f16* p = &Vt[(t * 16 + lo) * 72 + fb * 32 + hi * 8];
        v16h bV = make_v16(*(const v8h*)p, *(const v8h*)(p + 16));
        z = __builtin_amdgcn_wmma_f32_16x16x32_f16(false, aP[fb], false, bV,
                                                   (short)0, z, false, false);
      }
      acc[t] = z;
    }
    __syncthreads();
  }

  // normalize and write to [B,T,C] f16
  const int b = bh >> 2, h = bh & 3;
#pragma unroll
  for (int t = 0; t < 4; ++t) {
    int d = h * 64 + t * 16 + lo;
#pragma unroll
    for (int r = 0; r < 8; ++r) {
      int row = qrow0 + r + hi * 8;
      attn_ws[(size_t)(b * 2048 + row) * 256 + d] = (f16)(acc[t][r] / l_r[r]);
    }
  }
}

// ---------------------------------------------------------------------------
// Kernel 3: output projection.  out = attn @ w_proj^T + b_proj (f32 output).
// M=16384, N=256, K=256; same tiling as kernel 1, A already f16.
// ---------------------------------------------------------------------------
__global__ __launch_bounds__(256) void out_proj_kernel(
    const f16* __restrict__ attn_ws, const float* __restrict__ w_proj,
    const float* __restrict__ b_proj, float* __restrict__ out) {
  __shared__ f16 As[128 * 40];
  __shared__ f16 Bs[64 * 40];

  const int tid = threadIdx.x;
  const int lane = tid & 31;
  const int wv = tid >> 5;
  const int wm = wv & 3, wn = wv >> 2;
  const int lo = lane & 15, hi = lane >> 4;
  const int m_tile = blockIdx.y * 128;
  const int n_tile = blockIdx.x * 64;

  v8f acc[2][2] = {};

  const int ar = tid >> 1;            // 0..127
  const int ac = (tid & 1) * 16;      // 0 or 16

  for (int k0 = 0; k0 < 256; k0 += 32) {
    // A tile already f16: two 16B loads / thread
    {
      const f16* src = attn_ws + (size_t)(m_tile + ar) * 256 + k0 + ac;
      *(v8h*)(&As[ar * 40 + ac]) = *(const v8h*)src;
      *(v8h*)(&As[ar * 40 + ac + 8]) = *(const v8h*)(src + 8);
    }
    // B tile f32 -> f16, float4 loads
#pragma unroll
    for (int e = 0; e < 2; ++e) {
      int qidx = tid + 256 * e;
      int r = qidx >> 3, c4 = (qidx & 7) * 4;
      v4f f = *(const v4f*)&w_proj[(size_t)(n_tile + r) * 256 + k0 + c4];
      *(v4h*)&Bs[r * 40 + c4] = cvt4(f);
    }
    __syncthreads();

    v16h a[2], b[2];
#pragma unroll
    for (int i = 0; i < 2; ++i) {
      const f16* ap = &As[(wm * 32 + i * 16 + lo) * 40 + hi * 8];
      a[i] = make_v16(*(const v8h*)ap, *(const v8h*)(ap + 16));
    }
#pragma unroll
    for (int jn = 0; jn < 2; ++jn) {
      const f16* bp = &Bs[(wn * 32 + jn * 16 + lo) * 40 + hi * 8];
      b[jn] = make_v16(*(const v8h*)bp, *(const v8h*)(bp + 16));
    }
#pragma unroll
    for (int i = 0; i < 2; ++i)
#pragma unroll
      for (int jn = 0; jn < 2; ++jn)
        acc[i][jn] = __builtin_amdgcn_wmma_f32_16x16x32_f16(
            false, a[i], false, b[jn], (short)0, acc[i][jn], false, false);
    __syncthreads();
  }

#pragma unroll
  for (int jn = 0; jn < 2; ++jn) {
    int n = n_tile + wn * 32 + jn * 16 + lo;
    float bias = b_proj[n];
#pragma unroll
    for (int i = 0; i < 2; ++i) {
      int mrow = m_tile + wm * 32 + i * 16;
#pragma unroll
      for (int r = 0; r < 8; ++r) {
        int m = mrow + r + hi * 8;
        out[(size_t)m * 256 + n] = acc[i][jn][r] + bias;
      }
    }
  }
}

// ---------------------------------------------------------------------------
extern "C" void kernel_launch(void* const* d_in, const int* in_sizes, int n_in,
                              void* d_out, int out_size, void* d_ws,
                              size_t ws_size, hipStream_t stream) {
  const float* x = (const float*)d_in[0];
  const float* w_qkv = (const float*)d_in[1];
  const float* b_qkv = (const float*)d_in[2];
  const float* w_proj = (const float*)d_in[3];
  const float* b_proj = (const float*)d_in[4];
  float* out = (float*)d_out;

  char* ws = (char*)d_ws;
  f16* q_ws = (f16*)(ws);                          // 8 MB  [B,H,T,HD]
  f16* k_ws = (f16*)(ws + (size_t)(8u << 20));     // 8 MB
  f16* v_ws = (f16*)(ws + (size_t)(16u << 20));    // 8 MB
  f16* attn_ws = (f16*)(ws + (size_t)(24u << 20)); // 8 MB [B,T,C]

  // QKV projection: grid (N tiles = 768/64, M tiles = 16384/128)
  qkv_proj_kernel<<<dim3(12, 128), 256, 0, stream>>>(x, w_qkv, b_qkv, q_ws,
                                                     k_ws, v_ws);
  // Attention: grid (q-blocks = 2048/128, B*H = 32)
  attn_kernel<<<dim3(16, 32), 256, 0, stream>>>(q_ws, k_ws, v_ws, attn_ws);
  // Output projection: grid (N tiles = 256/64, M tiles)
  out_proj_kernel<<<dim3(4, 128), 256, 0, stream>>>(attn_ws, w_proj, b_proj,
                                                    out);
}
